// VectorQuantizer_42606075576662
// MI455X (gfx1250) — matrix operand order repaired and checked by reading, not compiled, not run on hardware
//
#include <hip/hip_runtime.h>

// Problem constants (from reference)
#define NPIX 8192      // 2*4*32*32 pixels
#define NEMB 8192
#define DIM  256
#define ZQ_ELEMS 2097152          // 2*256*4*32*32
#define LOSS_OFF ZQ_ELEMS
#define IDX_OFF (ZQ_ELEMS + 1)

#define KST 264        // padded LDS row stride (bf16 elems); 528B = 33*16B, rotates banks
#define KST_U32 132
#define ROW_B 528      // LDS row stride in bytes

typedef __attribute__((ext_vector_type(16))) __bf16 v16bf;
typedef __attribute__((ext_vector_type(8)))  __bf16 v8bf;
typedef __attribute__((ext_vector_type(8)))  float  v8f;

__device__ __forceinline__ v16bf frag16(const __bf16* rowBase, int e0, int e1) {
  v8bf a = *(const v8bf*)(rowBase + e0);
  v8bf b = *(const v8bf*)(rowBase + e1);
  v16bf r;
#pragma unroll
  for (int i = 0; i < 8; ++i) { r[i] = a[i]; r[i + 8] = b[i]; }
  return r;
}

__device__ __forceinline__ v8f wmma_bf16(v16bf a, v16bf b, v8f c) {
  return __builtin_amdgcn_wmma_f32_16x16x32_bf16(
      /*neg_a=*/false, a, /*neg_b=*/false, b,
      /*c_mod=*/(short)0, c, /*reuse_a=*/false, /*reuse_b=*/false);
}

// Async copy of one 64-row x 128-column(bf16) K-chunk: global -> LDS, b128 granularity.
// Global rows are 512B; chunk covers bytes [kChunk*256, kChunk*256+256) of each row.
__device__ __forceinline__ void async_tile_load(unsigned ldsBase, const __bf16* __restrict__ g,
                                                int rowBase, int kChunk, int tid) {
  const unsigned long long gbase = (unsigned long long)(uintptr_t)g;
#pragma unroll
  for (int it = 0; it < 8; ++it) {
    int idx = it * 128 + tid;          // 64 rows * 16 xfers = 1024 / 128 threads
    int r = idx >> 4, c = idx & 15;
    unsigned lds  = ldsBase + (unsigned)(r * ROW_B + kChunk * 256 + c * 16);
    unsigned goff = (unsigned)((rowBase + r) * 512 + kChunk * 256 + c * 16);
    asm volatile("global_load_async_to_lds_b128 %0, %1, %2"
                 :: "v"(lds), "v"(goff), "s"(gbase) : "memory");
  }
}

__device__ __forceinline__ void wait_async_barrier() {
  asm volatile("s_wait_asynccnt 0x0" ::: "memory");
  __syncthreads();
}

// ---------------- prep: z transpose (channel-first -> [p][d]) + bf16 hi/lo split
__global__ void prep_z(const float* __restrict__ z,
                       __bf16* __restrict__ zhi, __bf16* __restrict__ zlo) {
  __shared__ __align__(16) __bf16 tHi[32 * KST];
  __shared__ __align__(16) __bf16 tLo[32 * KST];
  const int tid = threadIdx.x;
  const int pBase = blockIdx.x * 32;
  // read: warp covers 32 consecutive pixels, same d -> coalesced on (h,w)
  for (int it = 0; it < 32; ++it) {
    int idx = it * 256 + tid;
    int pl = idx & 31, d = idx >> 5;
    int p = pBase + pl;
    float v = z[(p >> 12) * 1048576 + d * 4096 + (p & 4095)];
    __bf16 h = (__bf16)v;
    __bf16 l = (__bf16)(v - (float)h);
    tHi[pl * KST + d] = h;
    tLo[pl * KST + d] = l;
  }
  __syncthreads();
  // write: contiguous u32 rows -> coalesced
  unsigned* gh = (unsigned*)zhi;
  unsigned* gl = (unsigned*)zlo;
  const unsigned* sh = (const unsigned*)tHi;
  const unsigned* sl = (const unsigned*)tLo;
  for (int it = 0; it < 16; ++it) {
    int idx = it * 256 + tid;
    int r = idx >> 7, c = idx & 127;
    gh[(pBase + r) * 128 + c] = sh[r * KST_U32 + c];
    gl[(pBase + r) * 128 + c] = sl[r * KST_U32 + c];
  }
}

// ---------------- prep: codebook hi/lo split + squared norms
__global__ void prep_w(const float* __restrict__ w,
                       __bf16* __restrict__ whi, __bf16* __restrict__ wlo,
                       float* __restrict__ wnorm) {
  __shared__ float red[256];
  const int e = blockIdx.x, d = threadIdx.x;
  float v = w[e * DIM + d];
  __bf16 h = (__bf16)v;
  __bf16 l = (__bf16)(v - (float)h);
  whi[e * DIM + d] = h;
  wlo[e * DIM + d] = l;
  red[d] = v * v;
  __syncthreads();
  for (int s = 128; s > 0; s >>= 1) {
    if (d < s) red[d] += red[d + s];
    __syncthreads();
  }
  if (d == 0) wnorm[e] = red[0];
}

__global__ void init_min(unsigned long long* __restrict__ gmin) {
  int i = blockIdx.x * 256 + threadIdx.x;
  if (i < NPIX) gmin[i] = 0xFFFFFFFFFFFFFFFFULL;
}

// ---------------- fused GEMM (bf16x3) + per-row argmin
__device__ __forceinline__ void reduce_tile(v8f acc, int mBase, int nGlobBase,
                                            const float* __restrict__ wnorm,
                                            unsigned long long* rowMin, int lane) {
  const int l16 = lane & 15;
  const int n = nGlobBase + l16;
  const float wn = wnorm[n];
  const int rAdd = (lane >> 4) * 8;
#pragma unroll
  for (int r = 0; r < 8; ++r) {
    float d = wn - 2.0f * acc[r];
    unsigned u = __float_as_uint(d);
    unsigned hi = u ^ (unsigned)(((int)u >> 31) | (int)0x80000000);  // sortable key
    unsigned lo = (unsigned)n;
#pragma unroll
    for (int off = 1; off < 16; off <<= 1) {
      unsigned ohi = __shfl_xor(hi, off, 16);
      unsigned olo = __shfl_xor(lo, off, 16);
      if (ohi < hi || (ohi == hi && olo < lo)) { hi = ohi; lo = olo; }
    }
    if (l16 == 0) {
      unsigned long long pk = ((unsigned long long)hi << 32) | (unsigned long long)lo;
      atomicMin(&rowMin[mBase + rAdd + r], pk);
    }
  }
}

__device__ __forceinline__ void compute_chunk(int kBase, int kh,
    const __bf16* Ah0p, const __bf16* Al0p, const __bf16* Ah1p, const __bf16* Al1p,
    const __bf16* Bh0p, const __bf16* Bl0p, const __bf16* Bh1p, const __bf16* Bl1p,
    v8f& a00, v8f& a01, v8f& a10, v8f& a11) {
#pragma unroll
  for (int ks = 0; ks < 128; ks += 32) {
    const int k0 = kBase + ks;
    // A-frag (16x32 bf16): lanes<16 hold K {0..7,16..23}+k0, lanes>=16 hold {8..15,24..31}+k0
    const int aO0 = k0 + 8 * kh, aO1 = aO0 + 16;
    // B-frag (32x16 bf16): lanes<16 hold K {0..7,16..23}+k0, lanes>=16 hold {8..15,24..31}+k0
    const int bO0 = k0 + 8 * kh, bO1 = bO0 + 16;
    v16bf Ah0 = frag16(Ah0p, aO0, aO1);
    v16bf Al0 = frag16(Al0p, aO0, aO1);
    v16bf Ah1 = frag16(Ah1p, aO0, aO1);
    v16bf Al1 = frag16(Al1p, aO0, aO1);
    v16bf Bh0 = frag16(Bh0p, bO0, bO1);
    v16bf Bl0 = frag16(Bl0p, bO0, bO1);
    v16bf Bh1 = frag16(Bh1p, bO0, bO1);
    v16bf Bl1 = frag16(Bl1p, bO0, bO1);
    a00 = wmma_bf16(Ah0, Bh0, a00); a00 = wmma_bf16(Ah0, Bl0, a00); a00 = wmma_bf16(Al0, Bh0, a00);
    a01 = wmma_bf16(Ah0, Bh1, a01); a01 = wmma_bf16(Ah0, Bl1, a01); a01 = wmma_bf16(Al0, Bh1, a01);
    a10 = wmma_bf16(Ah1, Bh0, a10); a10 = wmma_bf16(Ah1, Bl0, a10); a10 = wmma_bf16(Al1, Bh0, a10);
    a11 = wmma_bf16(Ah1, Bh1, a11); a11 = wmma_bf16(Ah1, Bl1, a11); a11 = wmma_bf16(Al1, Bh1, a11);
  }
}

__global__ void __launch_bounds__(128)
gemm_argmin(const __bf16* __restrict__ zhi, const __bf16* __restrict__ zlo,
            const __bf16* __restrict__ whi, const __bf16* __restrict__ wlo,
            const float* __restrict__ wnorm, unsigned long long* __restrict__ gmin) {
  __shared__ __align__(16) __bf16 sZh[64 * KST];
  __shared__ __align__(16) __bf16 sZl[64 * KST];
  __shared__ __align__(16) __bf16 sWh[64 * KST];
  __shared__ __align__(16) __bf16 sWl[64 * KST];
  __shared__ unsigned long long rowMin[64];

  const int tid = threadIdx.x;
  const int lane = tid & 31;
  const int wv = tid >> 5;
  const int pBase = blockIdx.y * 64;
  const int nBase = blockIdx.x * 64;

  if (tid < 64) rowMin[tid] = 0xFFFFFFFFFFFFFFFFULL;

  const unsigned ldsZh = (unsigned)(uintptr_t)&sZh[0];
  const unsigned ldsZl = (unsigned)(uintptr_t)&sZl[0];
  const unsigned ldsWh = (unsigned)(uintptr_t)&sWh[0];
  const unsigned ldsWl = (unsigned)(uintptr_t)&sWl[0];

  // chunk 0 (K 0..127): async global -> LDS
  async_tile_load(ldsZh, zhi, pBase, 0, tid);
  async_tile_load(ldsZl, zlo, pBase, 0, tid);
  async_tile_load(ldsWh, whi, nBase, 0, tid);
  async_tile_load(ldsWl, wlo, nBase, 0, tid);
  wait_async_barrier();

  // issue chunk 1 (K 128..255) while computing chunk 0 (disjoint LDS bytes -> no WAR)
  async_tile_load(ldsZh, zhi, pBase, 1, tid);
  async_tile_load(ldsZl, zlo, pBase, 1, tid);
  async_tile_load(ldsWh, whi, nBase, 1, tid);
  async_tile_load(ldsWl, wlo, nBase, 1, tid);

  const int mHalf = wv >> 1, nHalf = wv & 1;
  const int l16 = lane & 15, kh = lane >> 4;

  const __bf16* Ah0p = sZh + (32 * mHalf + l16) * KST;
  const __bf16* Ah1p = Ah0p + 16 * KST;
  const __bf16* Al0p = sZl + (32 * mHalf + l16) * KST;
  const __bf16* Al1p = Al0p + 16 * KST;
  const __bf16* Bh0p = sWh + (32 * nHalf + l16) * KST;
  const __bf16* Bh1p = Bh0p + 16 * KST;
  const __bf16* Bl0p = sWl + (32 * nHalf + l16) * KST;
  const __bf16* Bl1p = Bl0p + 16 * KST;

  v8f a00 = {}, a01 = {}, a10 = {}, a11 = {};
  compute_chunk(0, kh, Ah0p, Al0p, Ah1p, Al1p, Bh0p, Bl0p, Bh1p, Bl1p, a00, a01, a10, a11);
  wait_async_barrier();
  compute_chunk(128, kh, Ah0p, Al0p, Ah1p, Al1p, Bh0p, Bl0p, Bh1p, Bl1p, a00, a01, a10, a11);

  reduce_tile(a00, 32 * mHalf + 0,  nBase + 32 * nHalf + 0,  wnorm, rowMin, lane);
  reduce_tile(a01, 32 * mHalf + 0,  nBase + 32 * nHalf + 16, wnorm, rowMin, lane);
  reduce_tile(a10, 32 * mHalf + 16, nBase + 32 * nHalf + 0,  wnorm, rowMin, lane);
  reduce_tile(a11, 32 * mHalf + 16, nBase + 32 * nHalf + 16, wnorm, rowMin, lane);
  __syncthreads();
  if (tid < 64) atomicMin(&gmin[pBase + tid], rowMin[tid]);
}

// ---------------- gather z_q, write index, per-block loss partials
__global__ void gather_out(const float* __restrict__ z, const float* __restrict__ emb,
                           const unsigned long long* __restrict__ gmin,
                           float* __restrict__ out, float* __restrict__ partial) {
  __shared__ float red[256];
  const int tid = threadIdx.x;
  const int gid = blockIdx.x * 256 + tid;   // flat channel-first index
  const int n = gid >> 20;
  const int rem = gid & 1048575;
  const int d = rem >> 12;
  const int phw = rem & 4095;
  const int p = (n << 12) | phw;
  const unsigned idx = (unsigned)(gmin[p] & 0xFFFFFFFFULL);
  const float zq = emb[idx * DIM + d];
  const float zl = z[gid];
  out[gid] = zq;                            // straight-through output == z_q
  if (d == 0) out[IDX_OFF + p] = (float)idx;
  float diff = zq - zl;
  red[tid] = diff * diff;
  __syncthreads();
  for (int s = 128; s > 0; s >>= 1) {
    if (tid < s) red[tid] += red[tid + s];
    __syncthreads();
  }
  if (tid == 0) partial[blockIdx.x] = red[0];
}

__global__ void loss_final(const float* __restrict__ partial, float* __restrict__ out) {
  __shared__ float red[256];
  const int tid = threadIdx.x;
  float s = 0.f;
  for (int i = tid; i < 8192; i += 256) s += partial[i];
  red[tid] = s;
  __syncthreads();
  for (int st = 128; st > 0; st >>= 1) {
    if (tid < st) red[tid] += red[tid + st];
    __syncthreads();
  }
  if (tid == 0) out[LOSS_OFF] = 1.25f * red[0] * (1.0f / (float)ZQ_ELEMS);
}

extern "C" void kernel_launch(void* const* d_in, const int* in_sizes, int n_in,
                              void* d_out, int out_size, void* d_ws, size_t ws_size,
                              hipStream_t stream) {
  (void)in_sizes; (void)n_in; (void)out_size; (void)ws_size;
  const float* z   = (const float*)d_in[0];
  const float* emb = (const float*)d_in[1];
  float* out = (float*)d_out;

  char* ws = (char*)d_ws;
  __bf16* zhi = (__bf16*)ws;
  __bf16* zlo = zhi + (size_t)NPIX * DIM;
  __bf16* whi = zlo + (size_t)NPIX * DIM;
  __bf16* wlo = whi + (size_t)NEMB * DIM;
  float*  wnorm = (float*)(wlo + (size_t)NEMB * DIM);
  unsigned long long* gmin = (unsigned long long*)(wnorm + NEMB);
  float* partial = (float*)(gmin + NPIX);

  init_min<<<32, 256, 0, stream>>>(gmin);
  prep_z<<<NPIX / 32, 256, 0, stream>>>(z, zhi, zlo);
  prep_w<<<NEMB, 256, 0, stream>>>(emb, whi, wlo, wnorm);
  gemm_argmin<<<dim3(NEMB / 64, NPIX / 64), 128, 0, stream>>>(zhi, zlo, whi, wlo, wnorm, gmin);
  gather_out<<<ZQ_ELEMS / 256, 256, 0, stream>>>(z, emb, gmin, out, partial);
  loss_final<<<1, 256, 0, stream>>>(partial, out);
}
